// GPT_28192165331555
// MI455X (gfx1250) — compile-verified, matrix operand order checked by reading, not run on hardware
//
#include <hip/hip_runtime.h>
#include <hip/hip_bf16.h>

// ---- model constants (match reference) ----
#define L_  6
#define D_  1024
#define H_  16
#define HS_ 64
#define RD_ 16
#define I_  4096
#define V_  50304
#define B_  2
#define T_  1024
#define M_  (B_*T_)

typedef __attribute__((ext_vector_type(16))) __bf16 v16bf;
typedef __attribute__((ext_vector_type(8)))  float  v8f;

union Frag { v16bf v; unsigned u[8]; };

// ---- CDNA5 async global->LDS path (guarded; falls back to sync copy) ----
#if defined(__has_builtin)
#if __has_builtin(__builtin_amdgcn_global_load_async_to_lds_b128)
#define HAVE_ASYNC_LDS 1
#endif
#endif
#ifndef HAVE_ASYNC_LDS
#define HAVE_ASYNC_LDS 0
#endif

#if HAVE_ASYNC_LDS
// builtin signature (from compiler diagnostic): param 1 is
// '__attribute__((vector_size(16))) int __device__ *'  (v4i32 in AS1)
typedef int v4i_vs __attribute__((vector_size(16)));
typedef __attribute__((address_space(1))) v4i_vs* as1_v4i;
typedef __attribute__((address_space(3))) v4i_vs* as3_v4i;
#endif

__device__ __forceinline__ void cp16_to_lds(const unsigned short* g, unsigned short* l) {
#if HAVE_ASYNC_LDS
  // flat shared pointer low 32 bits are the LDS byte offset (aperture rules, ISA 10.2)
  __builtin_amdgcn_global_load_async_to_lds_b128(
      (as1_v4i)(unsigned long long)g,
      (as3_v4i)(unsigned)(unsigned long long)l,
      0, 0);
#else
  *(uint4*)(l) = *(const uint4*)(g);
#endif
}
#if HAVE_ASYNC_LDS
#define WAIT_ASYNC_2() asm volatile("s_wait_asynccnt 0x2" ::: "memory")
#define WAIT_ASYNC_0() asm volatile("s_wait_asynccnt 0x0" ::: "memory")
#else
#define WAIT_ASYNC_2()
#define WAIT_ASYNC_0()
#endif

__device__ __forceinline__ unsigned short f2bf(float f) {
  unsigned x = __float_as_uint(f);
  unsigned r = (x + 0x7FFFu + ((x >> 16) & 1u)) >> 16;   // round-to-nearest-even
  return (unsigned short)r;
}

// K-pair base for WMMA 16-bit A/B fragment VGPR v, lane half hf (ISA 7.12.2)
__device__ __forceinline__ int frag_k(int v, int hf) {
  return ((v < 4) ? 2 * v : 16 + 2 * (v - 4)) + 8 * hf;
}

// ---------------- fp32 -> bf16 weight conversion ----------------
__global__ void cvt_f32_bf16(const float* __restrict__ src,
                             unsigned short* __restrict__ dst, long n) {
  long i = (long)blockIdx.x * blockDim.x + threadIdx.x;
  long stride = (long)gridDim.x * blockDim.x;
  for (; i < n; i += stride) dst[i] = f2bf(src[i]);
}

// ---------------- embedding gather ----------------
__global__ void embed_kernel(const int* __restrict__ ids,
                             const float* __restrict__ ew,
                             float* __restrict__ x) {
  int tok = blockIdx.x;
  int id = ids[tok];
  for (int i = threadIdx.x; i < D_; i += blockDim.x)
    x[(long)tok * D_ + i] = ew[(long)id * D_ + i];
}

// ---------------- layernorm: fp32 in, bf16 out ----------------
__global__ __launch_bounds__(256) void ln_kernel(const float* __restrict__ x,
                          const float* __restrict__ gamma,
                          const float* __restrict__ beta,
                          unsigned short* __restrict__ out) {
  __shared__ float red[2][8];
  int row = blockIdx.x;
  const float* xr = x + (long)row * D_;
  float v0[4];
  float sum = 0.f, sq = 0.f;
#pragma unroll
  for (int j = 0; j < 4; j++) {
    float f = xr[threadIdx.x + j * 256];
    v0[j] = f; sum += f; sq += f * f;
  }
#pragma unroll
  for (int off = 1; off < 32; off <<= 1) {
    sum += __shfl_xor(sum, off, 32);
    sq  += __shfl_xor(sq,  off, 32);
  }
  int wid = threadIdx.x >> 5;
  if ((threadIdx.x & 31) == 0) { red[0][wid] = sum; red[1][wid] = sq; }
  __syncthreads();
  sum = 0.f; sq = 0.f;
#pragma unroll
  for (int w = 0; w < 8; w++) { sum += red[0][w]; sq += red[1][w]; }
  float mean = sum * (1.0f / D_);
  float var  = sq * (1.0f / D_) - mean * mean;
  float inv  = rsqrtf(var + 1e-5f);
#pragma unroll
  for (int j = 0; j < 4; j++) {
    int idx = threadIdx.x + j * 256;
    out[(long)row * D_ + idx] = f2bf((v0[j] - mean) * inv * gamma[idx] + beta[idx]);
  }
}

// ---------------- generic bf16 WMMA GEMM (128x128x32, double-buffered) ----------------
// C[M,N] = A[M,K](bf16) * B[K,N](bf16) + bias (+gelu) (+res1) (+res2)
#define BM 128
#define BN 128
#define BK 32
#define APAD 56   // A LDS row stride: 112B (16B-aligned rows for b128 async, conflict-free frags)
#define BPAD 34   // transposed-B LDS row stride

__global__ __launch_bounds__(256) void gemm_bf16_kernel(
    const unsigned short* __restrict__ A, int lda,
    const unsigned short* __restrict__ Bw, int ldb,
    const float* __restrict__ bias,
    float* __restrict__ Cf, unsigned short* __restrict__ Cb,
    const float* __restrict__ res1, const float* __restrict__ res2,
    int N, int K, int do_gelu) {
  __shared__ __align__(16) unsigned short As[2][BM][APAD];
  __shared__ __align__(16) unsigned short Bs[2][BN][BPAD];

  const int tid  = threadIdx.x;
  const int lane = tid & 31;
  const int wid  = tid >> 5;
  const int wm   = wid & 3;     // 4 M-waves of 32 rows
  const int wn   = wid >> 2;    // 2 N-waves of 64 cols
  const int row0 = blockIdx.y * BM;
  const int col0 = blockIdx.x * BN;
  const int m16  = lane & 15;
  const int hf   = lane >> 4;

  v8f acc[2][4];
  v8f zero = {0.f, 0.f, 0.f, 0.f, 0.f, 0.f, 0.f, 0.f};
#pragma unroll
  for (int i = 0; i < 2; i++)
#pragma unroll
    for (int j = 0; j < 4; j++) acc[i][j] = zero;

  // A tile: 128x32, each thread copies two 16B chunks (row ar, k-offsets ak, ak+8)
  const int ar = tid >> 1;           // 0..127
  const int ak = (tid & 1) * 16;     // 0 or 16
  // B tile: 32x128, each thread loads 16 contiguous u16 of row bk, transposes to LDS
  const int bk  = tid >> 3;          // 0..31
  const int bn0 = (tid & 7) * 16;

  const unsigned short* Acur = A + (long)(row0 + ar) * lda + ak;
  const unsigned short* Bcur = Bw + (long)bk * ldb + col0 + bn0;
  const long bstep = (long)BK * ldb;

  // ---- prologue: tile 0 ----
  uint4 breg0 = *(const uint4*)(Bcur);
  uint4 breg1 = *(const uint4*)(Bcur + 8);
  cp16_to_lds(Acur,     &As[0][ar][ak]);
  cp16_to_lds(Acur + 8, &As[0][ar][ak + 8]);

  int p = 0;
  for (int kb = 0; kb < K; kb += BK, p ^= 1) {
    // transpose-store current B tile (regs -> Bs[p])
    {
      __align__(16) unsigned short tmp[16];
      *(uint4*)(tmp)     = breg0;
      *(uint4*)(tmp + 8) = breg1;
#pragma unroll
      for (int j = 0; j < 16; j++) Bs[p][bn0 + j][bk] = tmp[j];
    }
    // barrier 1: all waves done with previous compute (safe to overwrite buf p^1),
    // and this tile's B stores are LDS-visible.
    __syncthreads();

    const int more = (kb + BK < K);
    if (more) {
      // overlap next tile's data movement with this tile's WMMAs
      Bcur += bstep;
      breg0 = *(const uint4*)(Bcur);
      breg1 = *(const uint4*)(Bcur + 8);
      Acur += BK;
      cp16_to_lds(Acur,     &As[p ^ 1][ar][ak]);
      cp16_to_lds(Acur + 8, &As[p ^ 1][ar][ak + 8]);
      __builtin_prefetch(Bcur + bstep, 0, 1);          // global_prefetch_b8 (L2 warm)
      __builtin_prefetch(Acur + BK, 0, 1);
      WAIT_ASYNC_2();   // tile kb's two async copies complete (in-order); kb+BK's remain in flight
    } else {
      WAIT_ASYNC_0();
    }
    // barrier 2: every wave's async A-tile writes for tile kb are complete
    __syncthreads();

    Frag afr[2], bfr[4];
#pragma unroll
    for (int mt = 0; mt < 2; mt++)
#pragma unroll
      for (int v = 0; v < 8; v++)
        afr[mt].u[v] = *(const unsigned*)(&As[p][wm * 32 + mt * 16 + m16][frag_k(v, hf)]);
#pragma unroll
    for (int nt = 0; nt < 4; nt++)
#pragma unroll
      for (int v = 0; v < 8; v++)
        bfr[nt].u[v] = *(const unsigned*)(&Bs[p][wn * 64 + nt * 16 + m16][frag_k(v, hf)]);

#pragma unroll
    for (int mt = 0; mt < 2; mt++)
#pragma unroll
      for (int nt = 0; nt < 4; nt++)
        acc[mt][nt] = __builtin_amdgcn_wmma_f32_16x16x32_bf16(
            false, afr[mt].v, false, bfr[nt].v, (short)0, acc[mt][nt], false, false);
  }

  // epilogue
#pragma unroll
  for (int mt = 0; mt < 2; mt++)
#pragma unroll
    for (int nt = 0; nt < 4; nt++) {
      int col = col0 + wn * 64 + nt * 16 + m16;
#pragma unroll
      for (int r = 0; r < 8; r++) {
        int row = row0 + wm * 32 + mt * 16 + r + 8 * hf;
        float vv = acc[mt][nt][r];
        if (bias) vv += bias[col];
        if (do_gelu) vv = 0.5f * vv * (1.0f + erff(vv * 0.70710678118f));
        long idx = (long)row * N + col;
        if (res1) vv += res1[idx];
        if (res2) vv += res2[idx];
        if (Cf) Cf[idx] = vv;
        if (Cb) Cb[idx] = f2bf(vv);
      }
    }
}

// ---------------- rope + split to head-major bf16 Q/K/V ----------------
__global__ void rope_kernel(const float* __restrict__ qkv,
                            unsigned short* __restrict__ qb,
                            unsigned short* __restrict__ kb,
                            unsigned short* __restrict__ vb) {
  int pair = blockIdx.x * blockDim.x + threadIdx.x;   // token*H + h
  if (pair >= M_ * H_) return;
  int tok = pair / H_;
  int h   = pair % H_;
  int bb  = tok / T_;
  int t   = tok % T_;
  const float* base = qkv + (long)tok * (3 * D_) + h * (3 * HS_);
  long obase = ((long)(bb * H_ + h) * T_ + t) * HS_;

#pragma unroll
  for (int d = 0; d < 8; d++) {
    float invf = __powf(10000.0f, -(float)(2 * d) / (float)RD_);
    float a = (float)t * invf;
    float c = __cosf(a), s = __sinf(a);
    float q1 = base[d], q2 = base[d + 8];
    float k1 = base[HS_ + d], k2 = base[HS_ + d + 8];
    qb[obase + d]     = f2bf(q1 * c - q2 * s);
    qb[obase + d + 8] = f2bf(q2 * c + q1 * s);
    kb[obase + d]     = f2bf(k1 * c - k2 * s);
    kb[obase + d + 8] = f2bf(k2 * c + k1 * s);
  }
  for (int d = RD_; d < HS_; d++) {
    qb[obase + d] = f2bf(base[d]);
    kb[obase + d] = f2bf(base[HS_ + d]);
  }
  for (int d = 0; d < HS_; d++)
    vb[obase + d] = f2bf(base[2 * HS_ + d]);
}

// ---------------- flash-style attention (WMMA scores + WMMA P*V) ----------------
__global__ __launch_bounds__(256) void attn_kernel(
    const unsigned short* __restrict__ qb,
    const unsigned short* __restrict__ kkb,
    const unsigned short* __restrict__ vvb,
    unsigned short* __restrict__ ob) {     // [M_, D_] bf16, col = h*HS + d
  __shared__ __align__(16) unsigned short pbuf[8][16][BPAD];
  int bh   = blockIdx.x;
  int bb   = bh / H_;
  int h    = bh % H_;
  int lane = threadIdx.x & 31;
  int wid  = threadIdx.x >> 5;
  int q0   = blockIdx.y * 128 + wid * 16;
  int m16  = lane & 15;
  int hf   = lane >> 4;

  const unsigned short* Q  = qb  + ((long)bh * T_) * HS_;
  const unsigned short* Kk = kkb + ((long)bh * T_) * HS_;
  const unsigned short* Vv = vvb + ((long)bh * T_) * HS_;

  // Q fragments: 16x64 = two 16x32 A-frags (contiguous K pairs in global)
  Frag qa[2];
#pragma unroll
  for (int c = 0; c < 2; c++)
#pragma unroll
    for (int v = 0; v < 8; v++)
      qa[c].u[v] = *(const unsigned*)(Q + (long)(q0 + m16) * HS_ + frag_k(v, hf) + 32 * c);

  float mrow[8], lrow[8];
  v8f oacc[4];
  v8f zero = {0.f, 0.f, 0.f, 0.f, 0.f, 0.f, 0.f, 0.f};
#pragma unroll
  for (int r = 0; r < 8; r++) { mrow[r] = -1e30f; lrow[r] = 0.f; }
#pragma unroll
  for (int t = 0; t < 4; t++) oacc[t] = zero;

  for (int jb = 0; jb <= q0 + 15; jb += 32) {
    v8f s[2];
#pragma unroll
    for (int nt = 0; nt < 2; nt++) {
      Frag kf[2];
#pragma unroll
      for (int c = 0; c < 2; c++)
#pragma unroll
        for (int v = 0; v < 8; v++)
          kf[c].u[v] = *(const unsigned*)(Kk + (long)(jb + nt * 16 + m16) * HS_ +
                                          frag_k(v, hf) + 32 * c);
      v8f z = zero;
      z = __builtin_amdgcn_wmma_f32_16x16x32_bf16(false, qa[0].v, false, kf[0].v, (short)0, z, false, false);
      z = __builtin_amdgcn_wmma_f32_16x16x32_bf16(false, qa[1].v, false, kf[1].v, (short)0, z, false, false);
      s[nt] = z;
    }

    // scale + causal mask
#pragma unroll
    for (int nt = 0; nt < 2; nt++)
#pragma unroll
      for (int r = 0; r < 8; r++) {
        int qrow = q0 + r + 8 * hf;
        int kcol = jb + nt * 16 + m16;
        float sv = s[nt][r] * 0.125f;
        if (kcol > qrow) sv = -1e30f;
        s[nt][r] = sv;
      }

    // online softmax: row reductions across the 16-lane half (wave32-safe)
    float p0a[8], p1a[8];
#pragma unroll
    for (int r = 0; r < 8; r++) {
      float mx = fmaxf(s[0][r], s[1][r]);
#pragma unroll
      for (int off = 1; off < 16; off <<= 1) mx = fmaxf(mx, __shfl_xor(mx, off, 32));
      float mnew  = fmaxf(mrow[r], mx);
      float alpha = __expf(mrow[r] - mnew);
      float p0 = __expf(s[0][r] - mnew);
      float p1 = __expf(s[1][r] - mnew);
      float ps = p0 + p1;
#pragma unroll
      for (int off = 1; off < 16; off <<= 1) ps += __shfl_xor(ps, off, 32);
      lrow[r] = lrow[r] * alpha + ps;
      mrow[r] = mnew;
#pragma unroll
      for (int t = 0; t < 4; t++) oacc[t][r] *= alpha;
      p0a[r] = p0; p1a[r] = p1;
    }

    // repack P (C-layout) -> A-layout through wave-private LDS
#pragma unroll
    for (int r = 0; r < 8; r++) {
      int row = r + 8 * hf;
      pbuf[wid][row][m16]      = f2bf(p0a[r]);
      pbuf[wid][row][16 + m16] = f2bf(p1a[r]);
    }
    asm volatile("s_wait_dscnt 0" ::: "memory");
    Frag pf;
#pragma unroll
    for (int v = 0; v < 8; v++)
      pf.u[v] = *(const unsigned*)(&pbuf[wid][m16][frag_k(v, hf)]);

    // O += P(16x32) * V(32x64)
#pragma unroll
    for (int t = 0; t < 4; t++) {
      Frag vf;
#pragma unroll
      for (int v = 0; v < 8; v++) {
        int kk = frag_k(v, hf);
        int dimc = t * 16 + m16;
        unsigned lo = Vv[(long)(jb + kk) * HS_ + dimc];
        unsigned hi = Vv[(long)(jb + kk + 1) * HS_ + dimc];
        vf.u[v] = lo | (hi << 16);
      }
      oacc[t] = __builtin_amdgcn_wmma_f32_16x16x32_bf16(false, pf.v, false, vf.v, (short)0, oacc[t], false, false);
    }
  }

  // normalize + store [tok, h*HS + d] bf16
#pragma unroll
  for (int t = 0; t < 4; t++)
#pragma unroll
    for (int r = 0; r < 8; r++) {
      int qrow = q0 + r + 8 * hf;
      float vv = oacc[t][r] / lrow[r];
      long tok = (long)bb * T_ + qrow;
      ob[tok * D_ + h * HS_ + t * 16 + m16] = f2bf(vv);
    }
}

// ---------------- host orchestration ----------------
extern "C" void kernel_launch(void* const* d_in, const int* in_sizes, int n_in,
                              void* d_out, int out_size, void* d_ws, size_t ws_size,
                              hipStream_t stream) {
  (void)in_sizes; (void)n_in; (void)out_size; (void)ws_size;
  const int*   ids     = (const int*)d_in[0];
  const float* embed_w = (const float*)d_in[1];
  const float* ln1_s   = (const float*)d_in[2];
  const float* ln1_b   = (const float*)d_in[3];
  const float* ln2_s   = (const float*)d_in[4];
  const float* ln2_b   = (const float*)d_in[5];
  const float* qkv_w   = (const float*)d_in[6];
  const float* qkv_b   = (const float*)d_in[7];
  const float* dense_w = (const float*)d_in[8];
  const float* dense_b = (const float*)d_in[9];
  const float* fc1_w   = (const float*)d_in[10];
  const float* fc1_b   = (const float*)d_in[11];
  const float* fc2_w   = (const float*)d_in[12];
  const float* fc2_b   = (const float*)d_in[13];
  const float* fln_s   = (const float*)d_in[14];
  const float* fln_b   = (const float*)d_in[15];
  const float* head_w  = (const float*)d_in[16];

  char* ws = (char*)d_ws;
  size_t off = 0;
  auto carve = [&](size_t bytes) -> void* {
    void* p = ws + off;
    off += (bytes + 255) & ~(size_t)255;
    return p;
  };
  unsigned short* wqkv   = (unsigned short*)carve((size_t)L_ * D_ * 3 * D_ * 2);
  unsigned short* wdense = (unsigned short*)carve((size_t)L_ * D_ * D_ * 2);
  unsigned short* wfc1   = (unsigned short*)carve((size_t)L_ * D_ * I_ * 2);
  unsigned short* wfc2   = (unsigned short*)carve((size_t)L_ * I_ * D_ * 2);
  unsigned short* whead  = (unsigned short*)carve((size_t)D_ * V_ * 2);
  float*          x      = (float*)carve((size_t)M_ * D_ * 4);
  unsigned short* hbf    = (unsigned short*)carve((size_t)M_ * D_ * 2);
  float*          qkvf   = (float*)carve((size_t)M_ * 3 * D_ * 4);
  unsigned short* qbf    = (unsigned short*)carve((size_t)M_ * D_ * 2);
  unsigned short* kbf    = (unsigned short*)carve((size_t)M_ * D_ * 2);
  unsigned short* vbf    = (unsigned short*)carve((size_t)M_ * D_ * 2);
  unsigned short* attbf  = (unsigned short*)carve((size_t)M_ * D_ * 2);
  float*          attres = (float*)carve((size_t)M_ * D_ * 4);
  unsigned short* midbf  = (unsigned short*)carve((size_t)M_ * I_ * 2);

  // weights fp32 -> bf16 (one streaming pass, ~21us of HBM at 23.3 TB/s)
  cvt_f32_bf16<<<2048, 256, 0, stream>>>(qkv_w,   wqkv,   (long)L_ * D_ * 3 * D_);
  cvt_f32_bf16<<<2048, 256, 0, stream>>>(dense_w, wdense, (long)L_ * D_ * D_);
  cvt_f32_bf16<<<2048, 256, 0, stream>>>(fc1_w,   wfc1,   (long)L_ * D_ * I_);
  cvt_f32_bf16<<<2048, 256, 0, stream>>>(fc2_w,   wfc2,   (long)L_ * I_ * D_);
  cvt_f32_bf16<<<2048, 256, 0, stream>>>(head_w,  whead,  (long)D_ * V_);

  embed_kernel<<<M_, 256, 0, stream>>>(ids, embed_w, x);

  for (int l = 0; l < L_; l++) {
    ln_kernel<<<M_, 256, 0, stream>>>(x, ln1_s + l * D_, ln1_b + l * D_, hbf);
    gemm_bf16_kernel<<<dim3(3 * D_ / BN, M_ / BM), 256, 0, stream>>>(
        hbf, D_, wqkv + (size_t)l * D_ * 3 * D_, 3 * D_, qkv_b + (size_t)l * 3 * D_,
        qkvf, nullptr, nullptr, nullptr, 3 * D_, D_, 0);
    rope_kernel<<<(M_ * H_ + 255) / 256, 256, 0, stream>>>(qkvf, qbf, kbf, vbf);
    attn_kernel<<<dim3(B_ * H_, T_ / 128), 256, 0, stream>>>(qbf, kbf, vbf, attbf);
    gemm_bf16_kernel<<<dim3(D_ / BN, M_ / BM), 256, 0, stream>>>(
        attbf, D_, wdense + (size_t)l * D_ * D_, D_, dense_b + (size_t)l * D_,
        attres, nullptr, nullptr, nullptr, D_, D_, 0);
    ln_kernel<<<M_, 256, 0, stream>>>(x, ln2_s + l * D_, ln2_b + l * D_, hbf);
    gemm_bf16_kernel<<<dim3(I_ / BN, M_ / BM), 256, 0, stream>>>(
        hbf, D_, wfc1 + (size_t)l * D_ * I_, I_, fc1_b + (size_t)l * I_,
        nullptr, midbf, nullptr, nullptr, I_, D_, 1);
    // x = fc2(mid) + fc2_b + attres + x   (in-place residual: same thread reads then writes idx)
    gemm_bf16_kernel<<<dim3(D_ / BN, M_ / BM), 256, 0, stream>>>(
        midbf, I_, wfc2 + (size_t)l * I_ * D_, D_, fc2_b + (size_t)l * D_,
        x, nullptr, attres, x, D_, I_, 0);
  }

  ln_kernel<<<M_, 256, 0, stream>>>(x, fln_s, fln_b, hbf);
  gemm_bf16_kernel<<<dim3(V_ / BN, M_ / BM), 256, 0, stream>>>(
      hbf, D_, whead, V_, nullptr, (float*)d_out, nullptr, nullptr, nullptr, V_, D_, 0);
}
